// ImplicitRelationEncoder_25795573580568
// MI455X (gfx1250) — compile-verified, matrix operand order checked by reading, not run on hardware
//
#include <hip/hip_runtime.h>
#include <hip/hip_bf16.h>

// ---------------------------------------------------------------------------
// ImplicitRelationEncoder for MI455X (gfx1250, wave32, WMMA + TDM).
// Large contractions -> v_wmma_f32_16x16x32_bf16. GEMM K-loop is a double-
// buffered pipeline whose interior tiles are staged by the Tensor Data Mover
// (tensor_load_to_lds, TENSORcnt): one descriptor-driven DMA per operand tile,
// issued by wave 0, with hardware LDS row padding matching the banked layout.
// f32->bf16 packing happens at fragment-build time (v_cvt_pk_bf16_f32) and
// co-executes with the TRANS-class 16-bit WMMA ops. out_t einsum re-associated
// as att @ (feat @ Wout^T) to cut 21 GFLOP and ~800MB of traffic.
// ---------------------------------------------------------------------------

typedef __attribute__((ext_vector_type(16))) __bf16    v16bf;
typedef __attribute__((ext_vector_type(8)))  float     v8f;
typedef __attribute__((ext_vector_type(4)))  unsigned  v4u;
typedef __attribute__((ext_vector_type(8)))  unsigned  v8u;

#define BATCH   64
#define KSEQ    100
#define HEADS   16
#define DH      64
#define ROWS    (BATCH * KSEQ)          // 6400

__device__ __forceinline__ unsigned pack_bf16(float lo, float hi) {
    union { __hip_bfloat162 h; unsigned u; } c;
    c.h = __float22bfloat162_rn(float2{lo, hi});
    return c.u;
}

__device__ __forceinline__ float apply_act(float v, int act) {
    if (act == 1) return (v > 0.0f) ? v : 0.01f * v;              // leaky relu
    if (act == 2) return fmaxf(v, 0.0f);                          // relu
    if (act == 3) return __logf(fmaxf(v, 1e-6f));                 // log(max(relu,1e-6))
    return v;
}

// ---------------------------------------------------------------------------
// Tensor Data Mover helpers (CDNA5 ISA).
// D# group 1 (invariant per GEMM launch): data_size=4B, pad_enable, tile dims,
// global row stride.  pad_interval code: 4 -> 32 DWORDs, 5 -> 64 DWORDs;
// pad_amount code 1 -> 2 DWORDs (=> LDS row strides 34 / 66 floats).
// ---------------------------------------------------------------------------
__device__ __forceinline__ v8u tdm_group1(unsigned tile_d0, unsigned tile_d1,
                                          unsigned long long stride_elems,
                                          unsigned pad_code)
{
    const unsigned td0 = 0x7FFFFFFFu, td1 = 0x7FFFFFFFu;  // interior: no OOB clip
    v8u g;
    g[0] = (2u << 16) | (1u << 20) | (pad_code << 22) | (1u << 25);
    g[1] = (td0 & 0xFFFFu) << 16;                         // barrier_addr=0 | dim0.lo
    g[2] = (td0 >> 16) | ((td1 & 0xFFFFu) << 16);         // dim0.hi | dim1.lo
    g[3] = (td1 >> 16) | (tile_d0 << 16);                 // dim1.hi | tile_dim0
    g[4] = tile_d1 & 0xFFFFu;                             // tile_dim1 | tile_dim2=0
    g[5] = (unsigned)(stride_elems & 0xFFFFFFFFu);        // dim0_stride[31:0]
    g[6] = (unsigned)((stride_elems >> 32) & 0xFFFFu);    // dim0_stride[47:32]
    g[7] = 0u;                                            // dim1_stride=0
    return g;
}

// 2D tile load, groups 2/3 = NULL.  group0: count=1, lds_addr, 57b global
// address, type=2.  LDS byte addr = low 32 bits of generic shared pointer.
__device__ __forceinline__ void tdm_load_2d(unsigned lds_addr, const float* gptr,
                                            v8u g1)
{
    unsigned long long ga = (unsigned long long)gptr;
    v4u g0;
    g0[0] = 1u;                                           // count=1 (valid)
    g0[1] = lds_addr;
    g0[2] = (unsigned)ga;
    g0[3] = (unsigned)((ga >> 32) & 0x01FFFFFFu) | (2u << 30);   // type=2
    asm volatile("tensor_load_to_lds %0, %1" :: "s"(g0), "s"(g1) : "memory");
}

__device__ __forceinline__ void wait_tensor(int n) {      // n block-uniform
    switch (n) {
    case 0:  __builtin_amdgcn_s_wait_tensorcnt(0); break;
    case 1:  __builtin_amdgcn_s_wait_tensorcnt(1); break;
    default: __builtin_amdgcn_s_wait_tensorcnt(2); break;
    }
}

// ---------------------------------------------------------------------------
// Generic batched GEMM: C = act(A @ B + bias)
//   A: [M,K] (lda). B: [K,N] (ldb) or [N,K] if transB. C: [M,N] (ldc). f32.
//   Per-batch offsets: off = (z/div)*s1 + (z%div)*s2   (div >= 1)
// Block 256 = 8 waves; tile 128(M) x 64(N) x 32(K); wave = 32x32 -> 4 WMMA.
// Double-buffered f32 LDS tiles; interior tiles staged by TDM (wave 0 issues
// one tensor_load_to_lds per operand), edge tiles staged synchronously.
// ---------------------------------------------------------------------------
#define TM 128
#define TN 64
#define TK 32
#define LDS_AS 34          // A (and transB B) row stride in floats
#define LDS_BNT 66         // non-trans B row stride ([k][n] layout)
#define BS_FLOATS 2176     // max(64*34, 32*66)

struct StageArgs {
    const float* A; const float* B;
    int M, N, Kdim, lda, ldb, m0, n0, transB;
    bool aFastM, bFastN;
};

// stage one K-tile into (As, Bs); returns #TDM ops issued by wave 0 (uniform)
__device__ __forceinline__ int stage_tile(const StageArgs& s, int k0, int tid,
                                          float* As, float* Bs,
                                          v8u gA1, v8u gB1)
{
    const bool kin = (k0 + TK <= s.Kdim);
    const bool aT = s.aFastM && kin;
    const bool bT = s.bFastN && kin;
    if (tid < 32) {                      // wave 0 drives the TDM
        if (aT) tdm_load_2d((unsigned)(unsigned long long)As,
                            &s.A[(long long)s.m0 * s.lda + k0], gA1);
        if (bT) tdm_load_2d((unsigned)(unsigned long long)Bs,
                            s.transB ? &s.B[(long long)s.n0 * s.ldb + k0]
                                     : &s.B[(long long)k0 * s.ldb + s.n0], gB1);
    }
    if (!aT) {                           // edge A: scalar bounds-checked
        for (int i = tid; i < TM * TK; i += 256) {
            int mm = i >> 5, kk = i & 31;
            int gm = s.m0 + mm, gk = k0 + kk;
            As[mm * LDS_AS + kk] =
                (gm < s.M && gk < s.Kdim) ? s.A[(long long)gm * s.lda + gk] : 0.0f;
        }
    }
    if (!bT) {                           // edge B: scalar bounds-checked
        for (int i = tid; i < TN * TK; i += 256) {
            int nn = i >> 5, kk = i & 31;
            int gn = s.n0 + nn, gk = k0 + kk;
            float v = 0.0f;
            if (gn < s.N && gk < s.Kdim)
                v = s.transB ? s.B[(long long)gn * s.ldb + gk]
                             : s.B[(long long)gk * s.ldb + gn];
            if (s.transB) Bs[nn * LDS_AS + kk] = v;
            else          Bs[kk * LDS_BNT + nn] = v;
        }
    }
    return (aT ? 1 : 0) + (bT ? 1 : 0);
}

__global__ __launch_bounds__(256)
void gemm_bf16_wmma(const float* __restrict__ A, const float* __restrict__ B,
                    const float* __restrict__ bias, float* __restrict__ C,
                    int M, int N, int Kdim, int lda, int ldb, int ldc,
                    long long sA1, long long sA2, int divA,
                    long long sB1, long long sB2, int divB,
                    long long sC1, long long sC2, int divC,
                    int transB, int act)
{
    __shared__ float AsF[2][TM * LDS_AS];
    __shared__ float BsF[2][BS_FLOATS];

    const int z = blockIdx.z;
    A += (long long)(z / divA) * sA1 + (long long)(z % divA) * sA2;
    B += (long long)(z / divB) * sB1 + (long long)(z % divB) * sB2;
    C += (long long)(z / divC) * sC1 + (long long)(z % divC) * sC2;

    const int m0 = blockIdx.x * TM;
    const int n0 = blockIdx.y * TN;
    const int tid  = threadIdx.x;
    const int lane = tid & 31;
    const int wave = tid >> 5;
    const int wm = wave & 3;          // 0..3 -> 32 rows each
    const int wn = wave >> 2;         // 0..1 -> 32 cols each
    const int g  = lane >> 4;
    const int r  = lane & 15;

    StageArgs sa;
    sa.A = A; sa.B = B; sa.M = M; sa.N = N; sa.Kdim = Kdim;
    sa.lda = lda; sa.ldb = ldb; sa.m0 = m0; sa.n0 = n0; sa.transB = transB;
    sa.aFastM = (m0 + TM <= M);
    sa.bFastN = (n0 + TN <= N);

    // TDM group-1 descriptors (launch-invariant):
    //   A: rows of 32 floats (pad code 4), TM rows, stride lda
    //   B: transB -> rows of 32 floats (pad 4), TN rows, stride ldb
    //      else   -> rows of 64 floats (pad 5), TK rows, stride ldb
    const v8u gA1 = tdm_group1(TK, TM, (unsigned long long)lda, 4u);
    const v8u gB1 = transB ? tdm_group1(TK, TN, (unsigned long long)ldb, 4u)
                           : tdm_group1(TN, TK, (unsigned long long)ldb, 5u);

    v8f acc00 = {}, acc01 = {}, acc10 = {}, acc11 = {};

    int cur = 0;
    stage_tile(sa, 0, tid, AsF[0], BsF[0], gA1, gB1);    // prologue: tile 0

    for (int k0 = 0; k0 < Kdim; k0 += TK) {
        int nextIssued = 0;
        if (k0 + TK < Kdim)
            nextIssued = stage_tile(sa, k0 + TK, tid, AsF[cur ^ 1], BsF[cur ^ 1],
                                    gA1, gB1);
        wait_tensor(nextIssued);      // in-order TDM: drains tile-k0 transfers
        __syncthreads();

        const float* Ab = AsF[cur];
        const float* Bb = BsF[cur];
        union { v16bf v; unsigned u[8]; } fa0, fa1, fb0, fb1;
        #pragma unroll
        for (int p = 0; p < 8; ++p) {
            // VGPR p holds k-pair {kk,kk+1}; kk = 2p + 8g (p<4) else 16+2(p-4)+8g
            int kk = (p < 4) ? (2 * p + 8 * g) : (16 + 2 * (p - 4) + 8 * g);
            float2 a0 = *(const float2*)&Ab[(wm * 32 + r) * LDS_AS + kk];
            float2 a1 = *(const float2*)&Ab[(wm * 32 + 16 + r) * LDS_AS + kk];
            fa0.u[p] = pack_bf16(a0.x, a0.y);
            fa1.u[p] = pack_bf16(a1.x, a1.y);
            if (transB) {
                float2 b0 = *(const float2*)&Bb[(wn * 32 + r) * LDS_AS + kk];
                float2 b1 = *(const float2*)&Bb[(wn * 32 + 16 + r) * LDS_AS + kk];
                fb0.u[p] = pack_bf16(b0.x, b0.y);
                fb1.u[p] = pack_bf16(b1.x, b1.y);
            } else {
                int cA = wn * 32 + r, cB = wn * 32 + 16 + r;
                fb0.u[p] = pack_bf16(Bb[kk * LDS_BNT + cA], Bb[(kk + 1) * LDS_BNT + cA]);
                fb1.u[p] = pack_bf16(Bb[kk * LDS_BNT + cB], Bb[(kk + 1) * LDS_BNT + cB]);
            }
        }
        acc00 = __builtin_amdgcn_wmma_f32_16x16x32_bf16(false, fa0.v, false, fb0.v,
                                                        (short)0, acc00, false, false);
        acc01 = __builtin_amdgcn_wmma_f32_16x16x32_bf16(false, fa0.v, false, fb1.v,
                                                        (short)0, acc01, false, false);
        acc10 = __builtin_amdgcn_wmma_f32_16x16x32_bf16(false, fa1.v, false, fb0.v,
                                                        (short)0, acc10, false, false);
        acc11 = __builtin_amdgcn_wmma_f32_16x16x32_bf16(false, fa1.v, false, fb1.v,
                                                        (short)0, acc11, false, false);
        __syncthreads();              // readers done before buffer reuse
        cur ^= 1;
    }

    // epilogue: C/D layout -> VGPR rr, lane: row = rr + 8*g, col = lane&15
    const int c0 = n0 + wn * 32 + r;
    const int c1 = c0 + 16;
    const float bv0 = (bias != nullptr && c0 < N) ? bias[c0] : 0.0f;
    const float bv1 = (bias != nullptr && c1 < N) ? bias[c1] : 0.0f;
    #pragma unroll
    for (int rr = 0; rr < 8; ++rr) {
        int row0 = m0 + wm * 32 + 8 * g + rr;
        int row1 = row0 + 16;
        if (c0 < N) {
            if (row0 < M) C[(long long)row0 * ldc + c0] = apply_act(acc00[rr] + bv0, act);
            if (row1 < M) C[(long long)row1 * ldc + c0] = apply_act(acc10[rr] + bv0, act);
        }
        if (c1 < N) {
            if (row0 < M) C[(long long)row0 * ldc + c1] = apply_act(acc01[rr] + bv1, act);
            if (row1 < M) C[(long long)row1 * ldc + c1] = apply_act(acc11[rr] + bv1, act);
        }
    }
}

// ---------------------------------------------------------------------------
// Elementwise / reduction kernels
// ---------------------------------------------------------------------------
__global__ void build_vcat(const float* __restrict__ v, const float* __restrict__ b6,
                           float* __restrict__ vcat)
{
    long long i = (long long)blockIdx.x * 256 + threadIdx.x;
    const long long total = (long long)ROWS * 2054;
    if (i >= total) return;
    long long row = i / 2054; int c = (int)(i % 2054);
    vcat[i] = (c < 2048) ? v[row * 2048 + c] : b6[row * 6 + (c - 2048)];
}

__global__ __launch_bounds__(256)
void build_x(const float* __restrict__ cap, const float* __restrict__ q,
             const float* __restrict__ gamma, const float* __restrict__ beta,
             const float* __restrict__ mean,  const float* __restrict__ var,
             float* __restrict__ x)
{
    const int row = blockIdx.x;           // b*100 + k
    const int b   = row / KSEQ;
    const int t   = threadIdx.x;
    __shared__ float sm[256];
    float s = 0.0f;
    for (int d = t; d < 1024; d += 256) s += cap[(long long)row * 1024 + d];
    sm[t] = s; __syncthreads();
    for (int o = 128; o > 0; o >>= 1) { if (t < o) sm[t] += sm[t + o]; __syncthreads(); }
    const bool zq = (sm[0] == 0.0f);
    for (int d = t; d < 2048; d += 256) {
        float val = (d < 1024) ? cap[(long long)row * 1024 + d]
                               : (zq ? 0.0f : q[(long long)b * 1024 + (d - 1024)]);
        x[(long long)row * 2048 + d] =
            (val - mean[d]) * rsqrtf(var[d] + 1e-5f) * gamma[d] + beta[d];
    }
}

__global__ __launch_bounds__(256)
void rownorm_inv(const float* __restrict__ H, float* __restrict__ invn, int D)
{
    const int row = blockIdx.x; const int t = threadIdx.x;
    __shared__ float sm[256];
    float s = 0.0f;
    for (int d = t; d < D; d += 256) { float v = H[(long long)row * D + d]; s += v * v; }
    sm[t] = s; __syncthreads();
    for (int o = 128; o > 0; o >>= 1) { if (t < o) sm[t] += sm[t + o]; __syncthreads(); }
    if (t == 0) invn[row] = 1.0f / fmaxf(sqrtf(sm[0]), 1e-8f);
}

__global__ __launch_bounds__(128)
void cosine_adj_k(const float* __restrict__ H, const float* __restrict__ invn,
                  float* __restrict__ adj, int D)
{
    const int row = blockIdx.x;           // b*100 + k
    const int b   = row / KSEQ;
    const int j   = threadIdx.x;
    if (j >= KSEQ) return;
    const float* hr = H + (long long)row * D;
    const float* hj = H + (long long)(b * KSEQ + j) * D;
    float s = 0.0f;
    for (int d = 0; d < D; ++d) s += hr[d] * hj[d];
    adj[(long long)row * KSEQ + j] = s * invn[row] * invn[b * KSEQ + j];
}

// per-(b,k,h): aff + pos-log-bias + mask + v_bias, then softmax over n=100
__global__ __launch_bounds__(128)
void gat_attention(const float* __restrict__ qd, const float* __restrict__ kd,
                   const float* __restrict__ adj, const float* __restrict__ pb,
                   const float* __restrict__ wb, const float* __restrict__ bb,
                   float* __restrict__ att)
{
    const int zz = blockIdx.x;            // b*1600 + k*16 + h
    const int h = zz % HEADS;
    const int k = (zz / HEADS) % KSEQ;
    const int b = zz / (HEADS * KSEQ);
    const int t = threadIdx.x;
    __shared__ float sm[128];

    float aff = -3.0e38f;
    if (t < KSEQ) {
        const float* qrow = qd + (long long)(b * KSEQ + k) * 1024 + h * DH;
        const float* krow = kd + (long long)(b * KSEQ + t) * 1024 + h * DH;
        float s = 0.0f;
        #pragma unroll 8
        for (int d = 0; d < DH; ++d) s += qrow[d] * krow[d];
        s *= 0.125f;                                              // 1/sqrt(64)
        s += pb[((long long)(b * KSEQ + k) * KSEQ + t) * HEADS + h];
        const float av = adj[(long long)(b * KSEQ + k) * KSEQ + t];
        if (!(av > 0.0f)) s = -9.0e15f;
        s += av * wb[0] + bb[0];
        aff = s;
    }
    sm[t] = aff; __syncthreads();
    for (int o = 64; o > 0; o >>= 1) { if (t < o) sm[t] = fmaxf(sm[t], sm[t + o]); __syncthreads(); }
    const float mx = sm[0]; __syncthreads();
    const float e = (t < KSEQ) ? __expf(aff - mx) : 0.0f;
    sm[t] = e; __syncthreads();
    for (int o = 64; o > 0; o >>= 1) { if (t < o) sm[t] += sm[t + o]; __syncthreads(); }
    if (t < KSEQ)
        att[((long long)(b * KSEQ + k) * HEADS + h) * KSEQ + t] = e / sm[0];
}

__global__ void gat_final(const float* __restrict__ sf, const float* __restrict__ outb,
                          const float* __restrict__ bout, float* __restrict__ dst)
{
    long long i = (long long)blockIdx.x * 256 + threadIdx.x;
    if (i >= (long long)ROWS * 1024) return;
    int c = (int)(i % 1024);
    dst[i] = fmaxf(sf[i] + outb[i] + bout[c], 0.0f);
}

__global__ void ewmul(const float* __restrict__ a, const float* __restrict__ b,
                      float* __restrict__ c, long long n)
{
    long long i = (long long)blockIdx.x * 256 + threadIdx.x;
    if (i < n) c[i] = a[i] * b[i];
}

__global__ void resid_relu_writeout(const float* __restrict__ base, const float* __restrict__ t,
                                    float* __restrict__ out, int colOff)
{
    long long i = (long long)blockIdx.x * 256 + threadIdx.x;
    if (i >= (long long)ROWS * 1024) return;
    long long row = i / 1024; int c = (int)(i % 1024);
    out[row * 2048 + colOff + c] = base[i] + fmaxf(t[i], 0.0f);
}

// ---------------------------------------------------------------------------
// Host-side orchestration
// ---------------------------------------------------------------------------
static inline void launch_gemm(hipStream_t s, const float* A, const float* B,
                               const float* bias, float* C,
                               int M, int N, int K, int lda, int ldb, int ldc,
                               long long sA1, long long sA2, int divA,
                               long long sB1, long long sB2, int divB,
                               long long sC1, long long sC2, int divC,
                               int nbatch, int transB, int act)
{
    dim3 grid((M + TM - 1) / TM, (N + TN - 1) / TN, nbatch);
    gemm_bf16_wmma<<<grid, 256, 0, s>>>(A, B, bias, C, M, N, K, lda, ldb, ldc,
                                        sA1, sA2, divA, sB1, sB2, divB,
                                        sC1, sC2, divC, transB, act);
}

extern "C" void kernel_launch(void* const* d_in, const int* in_sizes, int n_in,
                              void* d_out, int out_size, void* d_ws, size_t ws_size,
                              hipStream_t stream)
{
    // inputs in setup_inputs() insertion order; params dict flattened in
    // insertion order (Wv..bn_var, gat_L{Ws..bb}, gat_R{Ws..bb}).
    const float* v    = (const float*)d_in[0];
    const float* q    = (const float*)d_in[1];
    const float* cap  = (const float*)d_in[2];
    const float* b6   = (const float*)d_in[3];
    const float* posL = (const float*)d_in[6];
    const float* posR = (const float*)d_in[7];
    const float* Wv = (const float*)d_in[8],  *bv = (const float*)d_in[9];
    const float* W1 = (const float*)d_in[10], *b1 = (const float*)d_in[11];
    const float* W2 = (const float*)d_in[12], *b2 = (const float*)d_in[13];
    const float* W3 = (const float*)d_in[14], *b3 = (const float*)d_in[15];
    const float* W4 = (const float*)d_in[16], *b4 = (const float*)d_in[17];
    const float* WfL = (const float*)d_in[18], *bfL = (const float*)d_in[19];
    const float* WfR = (const float*)d_in[20], *bfR = (const float*)d_in[21];
    const float* bn_g = (const float*)d_in[22], *bn_b = (const float*)d_in[23];
    const float* bn_m = (const float*)d_in[24], *bn_v = (const float*)d_in[25];
    const float* gL[12]; for (int i = 0; i < 12; ++i) gL[i] = (const float*)d_in[26 + i];
    const float* gR[12]; for (int i = 0; i < 12; ++i) gR[i] = (const float*)d_in[38 + i];
    // gat param order: 0 Ws,1 bs,2 Wq,3 bq,4 Wk,5 bk,6 Wp,7 bp,8 Wout,9 bout,10 wb,11 bb

    float* ws = (float*)d_ws;
    float* OUT = (float*)d_out;

    // workspace layout (f32 elements), with reuse:
    float* VCAT = ws;                         // 13,145,600 ; reused as PB later
    float* PB   = VCAT;                       // 10,240,000 (b,k,n,h)
    float* X    = ws + 13145600ll;            // 13,107,200 ; reused as ATT later
    float* ATT  = X;                          // 10,240,000 (b,k,h,n)
    float* IMPV = ws + 26252800ll;            //  6,553,600
    float* SF   = ws + 32806400ll;            //  6,553,600 (self_feat L -> capRel)
    float* T1   = ws + 39360000ll;            //  6,553,600
    float* T2   = ws + 45913600ll;            //  3,276,800
    float* T3   = ws + 49190400ll;            //  6,553,600 (self_feat R -> vRel)
    float* T4   = ws + 55744000ll;            //  6,553,600 (kd)
    float* KW   = ws + 62297600ll;            //  6,553,600 (b,h,n,o)
    float* ADJL = ws + 68851200ll;            //    640,000
    float* ADJR = ws + 69491200ll;            //    640,000
    float* CV   = ws + 70131200ll;            //  6,553,600
    float* NRM  = ws + 76684800ll;            //      6,400

    const long long EW = (long long)ROWS * 1024;
    const int EWB = (int)((EW + 255) / 256);

    // ---- front end -------------------------------------------------------
    {
        long long n = (long long)ROWS * 2054;
        build_vcat<<<(int)((n + 255) / 256), 256, 0, stream>>>(v, b6, VCAT);
    }
    launch_gemm(stream, VCAT, Wv, bv, IMPV, ROWS, 1024, 2054, 2054, 1024, 1024,
                0,0,1, 0,0,1, 0,0,1, 1, 0, 1);
    build_x<<<ROWS, 256, 0, stream>>>(cap, q, bn_g, bn_b, bn_m, bn_v, X);
    launch_gemm(stream, X,  W1, b1, T1, ROWS, 1024, 2048, 2048, 1024, 1024,
                0,0,1, 0,0,1, 0,0,1, 1, 0, 1);
    launch_gemm(stream, T1, W2, b2, T2, ROWS, 512, 1024, 1024, 512, 512,
                0,0,1, 0,0,1, 0,0,1, 1, 0, 1);
    rownorm_inv<<<ROWS, 256, 0, stream>>>(T2, NRM, 512);
    cosine_adj_k<<<ROWS, 128, 0, stream>>>(T2, NRM, ADJL, 512);
    launch_gemm(stream, IMPV, W3, b3, T1, ROWS, 1024, 1024, 1024, 1024, 1024,
                0,0,1, 0,0,1, 0,0,1, 1, 0, 1);
    launch_gemm(stream, T1, W4, b4, T2, ROWS, 512, 1024, 1024, 512, 512,
                0,0,1, 0,0,1, 0,0,1, 1, 0, 1);
    rownorm_inv<<<ROWS, 256, 0, stream>>>(T2, NRM, 512);
    cosine_adj_k<<<ROWS, 128, 0, stream>>>(T2, NRM, ADJR, 512);

    // ---- GAT (shared sub-pipeline, run twice) ----------------------------
    struct GatCfg { const float* const* p; const float* xin; int din;
                    const float* pos; const float* adj; float* sf; };
    GatCfg cfgs[2] = { { gL, X,    2048, posL, ADJL, SF },
                       { gR, IMPV, 1024, posR, ADJR, T3 } };
    for (int gi = 0; gi < 2; ++gi) {
        const GatCfg& c = cfgs[gi];
        launch_gemm(stream, c.xin, c.p[0], c.p[1], c.sf, ROWS, 1024, c.din,
                    c.din, 1024, 1024, 0,0,1, 0,0,1, 0,0,1, 1, 0, 0);
        launch_gemm(stream, c.sf, c.p[2], c.p[3], T1, ROWS, 1024, 1024,
                    1024, 1024, 1024, 0,0,1, 0,0,1, 0,0,1, 1, 0, 0);
        launch_gemm(stream, c.sf, c.p[4], c.p[5], T4, ROWS, 1024, 1024,
                    1024, 1024, 1024, 0,0,1, 0,0,1, 0,0,1, 1, 0, 0);
        // pos bias: log(max(pos@Wp+bp, 1e-6))   [M=640000, K=64, N=16]
        launch_gemm(stream, c.pos, c.p[6], c.p[7], PB, ROWS * KSEQ, HEADS, 64,
                    64, HEADS, HEADS, 0,0,1, 0,0,1, 0,0,1, 1, 0, 3);
        gat_attention<<<BATCH * KSEQ * HEADS, 128, 0, stream>>>(
            T1, T4, c.adj, PB, c.p[10], c.p[11], ATT);
        // KW[b,h] = feat[b] @ Wout[h]^T  (batched NT), batches z = b*16+h
        launch_gemm(stream, c.sf, c.p[8], nullptr, KW, KSEQ, DH, 1024,
                    1024, 1024, DH,
                    (long long)KSEQ * 1024, 0, HEADS,        // A: (z/16)*102400
                    0, (long long)DH * 1024, HEADS,          // B: (z%16)*65536, transB
                    (long long)KSEQ * DH, 0, 1,              // C: z*6400
                    BATCH * HEADS, 1, 0);
        // out[b,k,h*64+o] = att[b,:,h,:] @ KW[b,h]
        launch_gemm(stream, ATT, KW, nullptr, T1, KSEQ, DH, KSEQ,
                    HEADS * KSEQ, DH, 1024,
                    (long long)KSEQ * HEADS * KSEQ, KSEQ, HEADS,  // A: b*160000 + h*100
                    (long long)KSEQ * DH, 0, 1,                   // B: z*6400
                    (long long)KSEQ * 1024, DH, HEADS,            // C: b*102400 + h*64
                    BATCH * HEADS, 0, 0);
        gat_final<<<EWB, 256, 0, stream>>>(c.sf, T1, c.p[9], c.sf);
    }

    // ---- fusion + output -------------------------------------------------
    ewmul<<<EWB, 256, 0, stream>>>(T3, SF, CV, EW);   // cap_v = vRel * capRel
    launch_gemm(stream, CV, WfL, bfL, T1, ROWS, 1024, 1024, 1024, 1024, 1024,
                0,0,1, 0,0,1, 0,0,1, 1, 0, 1);
    resid_relu_writeout<<<EWB, 256, 0, stream>>>(IMPV, T1, OUT, 0);
    launch_gemm(stream, CV, WfR, bfR, T1, ROWS, 1024, 1024, 1024, 1024, 1024,
                0,0,1, 0,0,1, 0,0,1, 1, 0, 1);
    resid_relu_writeout<<<EWB, 256, 0, stream>>>(cap, T1, OUT, 1024);
}
